// BatchNorm2d_custom_180388626588
// MI455X (gfx1250) — compile-verified, matrix operand order checked by reading, not run on hardware
//
#include <hip/hip_runtime.h>
#include <math.h>

// Problem constants (B=64, C=256, H=56, W=56)
#define HW_      3136                 // 56*56
#define CHW_     802816               // 256*3136
#define NCH_     200704               // 64*3136 (elements per channel)
#define C_       256
#define SPLIT_   16                   // blocks per channel in reduction passes
#define TILES_   392                  // NCH_/512 wave-tiles per channel
#define WAVES_PER_CH_ (SPLIT_ * 8)    // 8 waves per 256-thread block

typedef __bf16 bf16_t;
typedef bf16_t v16bf __attribute__((ext_vector_type(16)));
typedef float  v8f   __attribute__((ext_vector_type(8)));

// Native RNE conversions -> v_cvt_pk_bf16_f32 / bf16->f32 widen
__device__ __forceinline__ bf16_t tobf(float x) { return (bf16_t)x; }
__device__ __forceinline__ float  qbf(float x)  { return (float)(bf16_t)x; }

__device__ __forceinline__ v16bf ones16() {
    v16bf o;
#pragma unroll
    for (int k = 0; k < 16; ++k)
        o[k] = __builtin_bit_cast(bf16_t, (unsigned short)0x3F80);  // bf16 1.0
    return o;
}

// -------- Pass 1: per-channel sum of q_bf16(x) via WMMA (A x ones) ----------
__global__ __launch_bounds__(256) void bn_sum_kernel(const float* __restrict__ inp,
                                                     float* __restrict__ partial) {
    const int c     = blockIdx.x >> 4;            // / SPLIT_
    const int split = blockIdx.x & (SPLIT_ - 1);
    const int lane  = threadIdx.x & 31;
    const int wave  = threadIdx.x >> 5;

    const v16bf onesM = ones16();
    v8f acc = {0.f, 0.f, 0.f, 0.f, 0.f, 0.f, 0.f, 0.f};
    const float* chan = inp + (size_t)c * HW_;

    for (int t = split * 8 + wave; t < TILES_; t += WAVES_PER_CH_) {
        const int o = t * 512 + lane * 16;        // element offset in channel space
        const int b = o / HW_;                    // 16-elem run never crosses a slab
        const int r = o - b * HW_;
        const float4* p = (const float4*)(chan + (size_t)b * CHW_ + r);

        float4 x0 = p[0], x1 = p[1], x2 = p[2], x3 = p[3];
        v16bf a;
        a[0]  = tobf(x0.x); a[1]  = tobf(x0.y); a[2]  = tobf(x0.z); a[3]  = tobf(x0.w);
        a[4]  = tobf(x1.x); a[5]  = tobf(x1.y); a[6]  = tobf(x1.z); a[7]  = tobf(x1.w);
        a[8]  = tobf(x2.x); a[9]  = tobf(x2.y); a[10] = tobf(x2.z); a[11] = tobf(x2.w);
        a[12] = tobf(x3.x); a[13] = tobf(x3.y); a[14] = tobf(x3.z); a[15] = tobf(x3.w);

        acc = __builtin_amdgcn_wmma_f32_16x16x32_bf16(false, a, false, onesM,
                                                      (short)0, acc, false, false);
    }

    // Every column of D holds the row sums -> sum everything, divide by 16.
    float s = acc[0] + acc[1] + acc[2] + acc[3] + acc[4] + acc[5] + acc[6] + acc[7];
#pragma unroll
    for (int off = 16; off >= 1; off >>= 1) s += __shfl_xor(s, off, 32);

    __shared__ float red[8];
    if (lane == 0) red[wave] = s;
    __syncthreads();
    if (threadIdx.x == 0) {
        float tsum = 0.f;
#pragma unroll
        for (int k = 0; k < 8; ++k) tsum += red[k];
        partial[c * SPLIT_ + split] = tsum * (1.0f / 16.0f);
    }
}

// -------- mean[c] = q_bf16(sum / n) -----------------------------------------
__global__ __launch_bounds__(256) void bn_mean_kernel(const float* __restrict__ partial,
                                                      float* __restrict__ meanArr) {
    const int c = threadIdx.x;
    float s = 0.f;
#pragma unroll
    for (int k = 0; k < SPLIT_; ++k) s += partial[c * SPLIT_ + k];
    meanArr[c] = qbf(s / 200704.0f);
}

// -------- Pass 2: per-channel sum of q_bf16((q_bf16(x)-mean)^2) via WMMA ----
__global__ __launch_bounds__(256) void bn_var_kernel(const float* __restrict__ inp,
                                                     const float* __restrict__ meanArr,
                                                     float* __restrict__ partial) {
    const int c     = blockIdx.x >> 4;
    const int split = blockIdx.x & (SPLIT_ - 1);
    const int lane  = threadIdx.x & 31;
    const int wave  = threadIdx.x >> 5;

    const float mean = meanArr[c];
    const v16bf onesM = ones16();
    v8f acc = {0.f, 0.f, 0.f, 0.f, 0.f, 0.f, 0.f, 0.f};
    const float* chan = inp + (size_t)c * HW_;

    for (int t = split * 8 + wave; t < TILES_; t += WAVES_PER_CH_) {
        const int o = t * 512 + lane * 16;
        const int b = o / HW_;
        const int r = o - b * HW_;
        const float4* p = (const float4*)(chan + (size_t)b * CHW_ + r);

        float xs[16];
        *(float4*)(xs + 0)  = p[0];
        *(float4*)(xs + 4)  = p[1];
        *(float4*)(xs + 8)  = p[2];
        *(float4*)(xs + 12) = p[3];

        v16bf a;
#pragma unroll
        for (int k = 0; k < 16; ++k) {
            float d = qbf(xs[k]) - mean;          // X(bf16) - avg, fp32
            a[k] = tobf(d * d);                   // q_bf16((X-avg)^2)
        }

        acc = __builtin_amdgcn_wmma_f32_16x16x32_bf16(false, a, false, onesM,
                                                      (short)0, acc, false, false);
    }

    float s = acc[0] + acc[1] + acc[2] + acc[3] + acc[4] + acc[5] + acc[6] + acc[7];
#pragma unroll
    for (int off = 16; off >= 1; off >>= 1) s += __shfl_xor(s, off, 32);

    __shared__ float red[8];
    if (lane == 0) red[wave] = s;
    __syncthreads();
    if (threadIdx.x == 0) {
        float tsum = 0.f;
#pragma unroll
        for (int k = 0; k < 8; ++k) tsum += red[k];
        partial[c * SPLIT_ + split] = tsum * (1.0f / 16.0f);
    }
}

// -------- var -> scale, quantized gamma -------------------------------------
__global__ __launch_bounds__(256) void bn_finalize_kernel(const float* __restrict__ partial,
                                                          const float* __restrict__ weight,
                                                          float* __restrict__ scaleArr,
                                                          float* __restrict__ gammaArr) {
    const int c = threadIdx.x;
    float vs = 0.f;
#pragma unroll
    for (int k = 0; k < SPLIT_; ++k) vs += partial[c * SPLIT_ + k];
    float var = qbf(vs);                 // q_bf16(sum(var_el))
    var = qbf(var / 200704.0f);          // q_bf16(var / n)
    scaleArr[c] = 1.0f / sqrtf(var + 1e-5f);
    gammaArr[c] = qbf(weight[c]);
}

// -------- Pass 3: streaming normalize, 1 float4 per thread ------------------
__global__ __launch_bounds__(256) void bn_norm_kernel(const float* __restrict__ inp,
                                                      const float* __restrict__ meanArr,
                                                      const float* __restrict__ scaleArr,
                                                      const float* __restrict__ gammaArr,
                                                      const float* __restrict__ beta,
                                                      float* __restrict__ out) {
    const int i = blockIdx.x * 256 + threadIdx.x;     // float4 index; grid covers exactly
    const int c = (i / 784) & 255;                    // 784 = HW/4, C=256 pow2

    const float mean = meanArr[c];
    const float sc   = scaleArr[c];
    const float g    = gammaArr[c];
    const float bt   = beta[c];

    float4 x = ((const float4*)inp)[i];
    float4 y;
    y.x = qbf(qbf(qbf((qbf(x.x) - mean) * sc) * g) + bt);
    y.y = qbf(qbf(qbf((qbf(x.y) - mean) * sc) * g) + bt);
    y.z = qbf(qbf(qbf((qbf(x.z) - mean) * sc) * g) + bt);
    y.w = qbf(qbf(qbf((qbf(x.w) - mean) * sc) * g) + bt);
    ((float4*)out)[i] = y;
}

extern "C" void kernel_launch(void* const* d_in, const int* in_sizes, int n_in,
                              void* d_out, int out_size, void* d_ws, size_t ws_size,
                              hipStream_t stream) {
    (void)in_sizes; (void)n_in; (void)out_size; (void)ws_size;
    const float* inp    = (const float*)d_in[0];
    const float* weight = (const float*)d_in[1];
    const float* beta   = (const float*)d_in[2];
    float* out = (float*)d_out;

    float* ws       = (float*)d_ws;
    float* partial  = ws;                 // [0, 4096)   per-(c,split) partials (reused)
    float* meanArr  = ws + 4096;          // [4096, 4352)
    float* scaleArr = ws + 4352;          // [4352, 4608)
    float* gammaArr = ws + 4608;          // [4608, 4864)

    bn_sum_kernel     <<<C_ * SPLIT_, 256, 0, stream>>>(inp, partial);
    bn_mean_kernel    <<<1,           256, 0, stream>>>(partial, meanArr);
    bn_var_kernel     <<<C_ * SPLIT_, 256, 0, stream>>>(inp, meanArr, partial);
    bn_finalize_kernel<<<1,           256, 0, stream>>>(partial, weight, scaleArr, gammaArr);
    bn_norm_kernel    <<<12845056 / 256, 256, 0, stream>>>(inp, meanArr, scaleArr, gammaArr,
                                                           beta, out);
}
